// ValueCritic_53961969107285
// MI455X (gfx1250) — matrix operand order repaired and checked by reading, not compiled, not run on hardware
//
#include <hip/hip_runtime.h>
#include <hip/hip_bf16.h>
#include <math.h>

// ---------------------------------------------------------------------------
// ValueCritic transformer, fused per-batch-element, f16 WMMA / f32 accumulate.
// Shapes: B=4096, L=64, T=256, H=8, HD=32, NL=4, DFF=1024, DIN=6
// ---------------------------------------------------------------------------

#define B_   4096
#define L_   64
#define T_   256
#define H_   8
#define HD_  32
#define NL_  4
#define DFF_ 1024
#define DIN_ 6

typedef __attribute__((ext_vector_type(16))) _Float16 v16h;
typedef __attribute__((ext_vector_type(8)))  _Float16 v8h;
typedef __attribute__((ext_vector_type(8)))  float    v8f;

// ------------------------- WMMA helpers (wave32) ---------------------------
// D = A(16x32 f16) * B(32x16 f16) + C(16x16 f32)
__device__ __forceinline__ v8f wmma_f16(v16h a, v16h b, v8f c) {
  return __builtin_amdgcn_wmma_f32_16x16x32_f16(
      /*neg_a=*/false, a, /*neg_b=*/false, b,
      /*c_mod=*/(short)0, c, /*reuse_a=*/false, /*reuse_b=*/false);
}

__device__ __forceinline__ v16h make16(v8h lo, v8h hi) {
  return __builtin_shufflevector(lo, hi, 0, 1, 2, 3, 4, 5, 6, 7,
                                 8, 9, 10, 11, 12, 13, 14, 15);
}

// A tile 16x32 (MxK), row-major source with leading dim lda (halves).
// ISA layout: lane<16 -> M=lane, K={0..7}U{16..23}; lane>=16 -> K={8..15}U{24..31}
// All call sites are 16-byte aligned -> two 128-bit loads.
__device__ __forceinline__ v16h load_A16x32(const _Float16* __restrict__ p,
                                            int lda, int lane) {
  int m  = lane & 15;
  int k0 = (lane >> 4) << 3;     // 0 or 8
  const _Float16* r = p + (size_t)m * lda + k0;
  v8h lo = *(const v8h*)(r);
  v8h hi = *(const v8h*)(r + 16);
  return make16(lo, hi);
}

// B tile 32x16 (KxN) from an N-major (row = n, K contiguous) source, ld=ldw.
// lane<16 -> N=lane, K={0..7}U{16..23} ; lane>=16 -> N=lane-16, K={8..15}U{24..31}
__device__ __forceinline__ v16h load_B_NK(const _Float16* __restrict__ w,
                                          int ldw, int lane) {
  int n  = lane & 15;
  int k0 = (lane >> 4) << 3;     // 0 or 8
  const _Float16* r = w + (size_t)n * ldw + k0;
  v8h lo = *(const v8h*)(r);
  v8h hi = *(const v8h*)(r + 16);
  return make16(lo, hi);
}

// ------------------------- weight conversion -------------------------------
__global__ void cvt_f16_kernel(const float* __restrict__ s,
                               _Float16* __restrict__ d, int n) {
  int i = blockIdx.x * blockDim.x + threadIdx.x;
  if (i < n) d[i] = (_Float16)s[i];
}

// link_W2 (L,T,T) -> transposed f16: W2t[l][s][t] = W2[l][t][s]
__global__ void cvt_w2t_kernel(const float* __restrict__ src,
                               _Float16* __restrict__ dst) {
  int l = blockIdx.y;
  int i = blockIdx.x * 256 + threadIdx.x;   // over T*T
  int t = i >> 8, s2 = i & 255;
  dst[((size_t)l * T_ + s2) * T_ + t] =
      (_Float16)src[((size_t)l * T_ + t) * T_ + s2];
}

// ------------------------- input normalization -----------------------------
// grid = B, block = 64 (one thread per link)
__global__ void norm_kernel(const float* __restrict__ ls,
                            float* __restrict__ xout) {
  int b = blockIdx.x, l = threadIdx.x;
  const float* r = ls + ((size_t)b * L_ + l) * DIN_;
  float x0 = r[0], x1 = r[1], x2 = r[2], x3 = r[3], x4 = r[4], x5 = r[5];
  float p  = fmaxf(x0, 0.f);
  float tx = fmaxf(x1, 0.f);

  __shared__ float s[L_];
  __shared__ float tref_s;
  s[l] = p + tx;
  __syncthreads();
  // rank-select the 32nd smallest (== sort()[31]) with stable tie-break
  float sv = s[l];
  int rank = 0;
  for (int j = 0; j < L_; ++j) {
    float o = s[j];
    rank += (o < sv) || (o == sv && j < l);
  }
  if (rank == (L_ - 1) / 2) tref_s = fmaxf(sv, 1e-6f);
  __syncthreads();
  float t = tref_s;

  float f[5];
  f[0] = p / t;
  f[1] = tx / t;
  f[2] = x2;
  f[3] = log1pf(fmaxf(x3, 0.f));
  f[4] = log1pf(fmaxf(x4, 0.f) / t);

  __shared__ float fs[5][L_];
  __shared__ float mn[5], sd[5];
#pragma unroll
  for (int i = 0; i < 5; ++i) fs[i][l] = f[i];
  __syncthreads();
  if (l < 5) {
    float m = 0.f;
    for (int j = 0; j < L_; ++j) m += fs[l][j];
    m *= (1.f / L_);
    float v = 0.f;
    for (int j = 0; j < L_; ++j) { float d = fs[l][j] - m; v += d * d; }
    v *= (1.f / L_);
    mn[l] = m; sd[l] = sqrtf(v) + 1e-9f;
  }
  __syncthreads();
#pragma unroll
  for (int i = 0; i < 5; ++i) {
    float lo = mn[i] - 3.f * sd[i], hi = mn[i] + 3.f * sd[i];
    f[i] = fminf(fmaxf(f[i], lo), hi);
  }
  __syncthreads();            // everyone done reading old mn/sd
#pragma unroll
  for (int i = 0; i < 5; ++i) fs[i][l] = f[i];
  __syncthreads();
  if (l < 5) {
    float m = 0.f;
    for (int j = 0; j < L_; ++j) m += fs[l][j];
    m *= (1.f / L_);
    float v = 0.f;
    for (int j = 0; j < L_; ++j) { float d = fs[l][j] - m; v += d * d; }
    v *= (1.f / L_);
    mn[l] = m; sd[l] = sqrtf(v) + 1e-9f;
  }
  __syncthreads();
  float* o = xout + ((size_t)b * L_ + l) * DIN_;
#pragma unroll
  for (int i = 0; i < 5; ++i) o[i] = (f[i] - mn[i]) / sd[i];
  o[5] = log1pf(fmaxf(x5, 0.f));
}

// ------------------------- link embedding + posenc -------------------------
__device__ __forceinline__ float posenc(int pos, int n) {
  int j = n >> 1;
  float div = expf(-logf(10000.f) * (2.f * j) / (float)T_);
  float a = (float)pos * div;
  return (n & 1) ? cosf(a) : sinf(a);
}

// grid = (B/64, L); block = 256.  y[b,l,:] = relu(x@W1[l]+b1[l])@W2[l]+b2[l]+pe[l]
__global__ void link_kernel(const float* __restrict__ x,
                            const float* __restrict__ W1,
                            const float* __restrict__ b1,
                            const _Float16* __restrict__ W2t,   // (L, s, t) N-major
                            const float* __restrict__ b2,
                            float* __restrict__ y) {
  int bt = blockIdx.x, l = blockIdx.y;
  int b0 = bt * 64;
  int tid = threadIdx.x, w = tid >> 5, lane = tid & 31;

  __shared__ float    xs[64][DIN_];
  __shared__ _Float16 hs[64 * T_];

  for (int i = tid; i < 64 * DIN_; i += 256) {
    int row = i / DIN_, d = i % DIN_;
    xs[row][d] = x[((size_t)(b0 + row) * L_ + l) * DIN_ + d];
  }
  __syncthreads();

  // h = relu(x @ W1[l] + b1[l]) : K=6 done on VALU, one column per thread
  {
    const float* w1 = W1 + (size_t)l * DIN_ * T_;
    float bv = b1[(size_t)l * T_ + tid];
    float wcol[DIN_];
#pragma unroll
    for (int d = 0; d < DIN_; ++d) wcol[d] = w1[d * T_ + tid];
    for (int i = 0; i < 64; ++i) {
      float acc = bv;
#pragma unroll
      for (int d = 0; d < DIN_; ++d) acc += xs[i][d] * wcol[d];
      hs[i * T_ + tid] = (_Float16)fmaxf(acc, 0.f);
    }
  }
  __syncthreads();

  // tok = h @ W2[l] ; B from transposed weights (N-major, contiguous K)
  const _Float16* w2 = W2t + (size_t)l * T_ * T_;
  for (int t = w; t < 64; t += 8) {
    int ti = t >> 4, tj = t & 15;
    v8f c = {};
#pragma unroll
    for (int kk = 0; kk < 8; ++kk) {
      v16h a  = load_A16x32(hs + ti * 16 * T_ + kk * 32, T_, lane);
      v16h bm = load_B_NK(w2 + (size_t)(tj * 16) * T_ + kk * 32, T_, lane);
      c = wmma_f16(a, bm, c);
    }
    int n = lane & 15, m0 = (lane >> 4) << 3;
    int col = tj * 16 + n;
    float bias = b2[(size_t)l * T_ + col] + posenc(l, col);
#pragma unroll
    for (int r = 0; r < 8; ++r) {
      int row = ti * 16 + m0 + r;
      y[((size_t)(b0 + row) * L_ + l) * T_ + col] = c[r] + bias;
    }
  }
}

// ------------------------- fused transformer layer -------------------------
// grid = B, block = 256 (8 waves). Dynamic LDS = 286720 B (<= 320KB/WGP).
#define LAYER_SMEM (65536 + 32768 + 188416)

__global__ void layer_kernel(float* __restrict__ y,
                             const _Float16* __restrict__ Wqkv, const float* __restrict__ bqkv,
                             const _Float16* __restrict__ Wo,   const float* __restrict__ bo,
                             const float* __restrict__ g1, const float* __restrict__ be1,
                             const _Float16* __restrict__ Wf1,  const float* __restrict__ bf1,
                             const _Float16* __restrict__ Wf2,  const float* __restrict__ bf2,
                             const float* __restrict__ g2, const float* __restrict__ be2) {
  extern __shared__ char smem[];
  float*    y32 = (float*)smem;                              // 64x256 f32  (64 KB)
  _Float16* yh  = (_Float16*)(smem + 65536);                 // 64x256 f16  (32 KB)
  char*     arena = smem + 65536 + 32768;                    // 184 KB arena
  _Float16* qkv = (_Float16*)arena;                          // 64x768 f16  (96 KB)
  float*    Sf  = (float*)(arena + 98304);                   // 64x64  f32  (16 KB)
  _Float16* Sh  = (_Float16*)(arena + 114688);               // 64x64  f16  ( 8 KB)
  _Float16* ao  = (_Float16*)(arena + 122880);               // 64x256 f16  (32 KB)
  _Float16* vtr = (_Float16*)(arena + 155648);               // 256x64 f16  (32 KB) V^T
  _Float16* ffh = (_Float16*)arena;                          // 64x1024 f16 (128 KB, reuses arena)

  int b = blockIdx.x, tid = threadIdx.x, w = tid >> 5, lane = tid & 31;
  float* yg = y + (size_t)b * L_ * T_;

  // ---- load residual master + f16 copy ----
  for (int i = tid; i < L_ * T_; i += 256) {
    float v = yg[i];
    y32[i] = v;
    yh[i]  = (_Float16)v;
  }
  __syncthreads();

  // ---- QKV: (64x768) = yh(64x256) @ Wqkv^T ; Wqkv is (768,256) N-major ----
  // V columns (col>=512) are additionally written transposed into vtr[d][token].
  for (int t = w; t < 192; t += 8) {
    int ti = t / 48, tj = t % 48;
    v8f c = {};
#pragma unroll
    for (int kk = 0; kk < 8; ++kk) {
      v16h a  = load_A16x32(yh + ti * 16 * T_ + kk * 32, T_, lane);
      v16h bm = load_B_NK(Wqkv + (size_t)(tj * 16) * T_ + kk * 32, T_, lane);
      c = wmma_f16(a, bm, c);
    }
    int n = lane & 15, m0 = (lane >> 4) << 3;
    int col = tj * 16 + n;
    float bias = bqkv[col];
    if (tj >= 32) {            // V tile (wave-uniform branch)
#pragma unroll
      for (int r = 0; r < 8; ++r) {
        int row = ti * 16 + m0 + r;
        _Float16 v = (_Float16)(c[r] + bias);
        qkv[(size_t)row * 768 + col] = v;
        vtr[(size_t)(col - 512) * 64 + row] = v;
      }
    } else {
#pragma unroll
      for (int r = 0; r < 8; ++r)
        qkv[(size_t)(ti * 16 + m0 + r) * 768 + col] = (_Float16)(c[r] + bias);
    }
  }
  __syncthreads();

  // ---- attention: heads sequential, all 8 waves per head ----
  const float scale = 0.17677669529663687f;  // 1/sqrt(32)
  for (int h = 0; h < H_; ++h) {
    // scores S = Q K^T * scale   (64x64, K=32 -> single WMMA per tile)
    for (int t = w; t < 16; t += 8) {
      int ti = t >> 2, tj = t & 3;
      v16h a  = load_A16x32(qkv + (size_t)(ti * 16) * 768 + h * HD_, 768, lane);
      v16h bm = load_B_NK(qkv + (size_t)(tj * 16) * 768 + T_ + h * HD_, 768, lane);
      v8f c = {};
      c = wmma_f16(a, bm, c);
      int n = lane & 15, m0 = (lane >> 4) << 3;
#pragma unroll
      for (int r = 0; r < 8; ++r)
        Sf[(ti * 16 + m0 + r) * 64 + tj * 16 + n] = c[r] * scale;
    }
    __syncthreads();
    // row softmax -> f16
    if (tid < 64) {
      float* row = Sf + tid * 64;
      float mx = -3.4e38f;
      for (int j = 0; j < 64; ++j) mx = fmaxf(mx, row[j]);
      float s = 0.f;
      for (int j = 0; j < 64; ++j) s += expf(row[j] - mx);
      float inv = 1.f / s;
      for (int j = 0; j < 64; ++j)
        Sh[tid * 64 + j] = (_Float16)(expf(row[j] - mx) * inv);
    }
    __syncthreads();
    // out_h = softmax @ V : B from vtr (N-major, contiguous K over tokens)
    for (int t = w; t < 8; t += 8) {
      int ti = t >> 1, tj = t & 1;
      v8f c = {};
#pragma unroll
      for (int kk = 0; kk < 2; ++kk) {
        v16h a  = load_A16x32(Sh + ti * 16 * 64 + kk * 32, 64, lane);
        v16h bm = load_B_NK(vtr + (size_t)(h * HD_ + tj * 16) * 64 + kk * 32,
                            64, lane);
        c = wmma_f16(a, bm, c);
      }
      int n = lane & 15, m0 = (lane >> 4) << 3;
#pragma unroll
      for (int r = 0; r < 8; ++r)
        ao[(size_t)(ti * 16 + m0 + r) * T_ + h * HD_ + tj * 16 + n] = (_Float16)c[r];
    }
    __syncthreads();
  }

  // ---- O projection + residual ----
  for (int t = w; t < 64; t += 8) {
    int ti = t >> 4, tj = t & 15;
    v8f c = {};
#pragma unroll
    for (int kk = 0; kk < 8; ++kk) {
      v16h a  = load_A16x32(ao + ti * 16 * T_ + kk * 32, T_, lane);
      v16h bm = load_B_NK(Wo + (size_t)(tj * 16) * T_ + kk * 32, T_, lane);
      c = wmma_f16(a, bm, c);
    }
    int n = lane & 15, m0 = (lane >> 4) << 3;
    int col = tj * 16 + n;
    float bias = bo[col];
#pragma unroll
    for (int r = 0; r < 8; ++r)
      y32[(ti * 16 + m0 + r) * T_ + col] += c[r] + bias;
  }
  __syncthreads();

  // ---- LayerNorm 1 (one row per thread) ----
  if (tid < 64) {
    float* row = y32 + tid * T_;
    float m = 0.f;
    for (int j = 0; j < T_; ++j) m += row[j];
    m *= (1.f / T_);
    float v = 0.f;
    for (int j = 0; j < T_; ++j) { float d = row[j] - m; v += d * d; }
    v *= (1.f / T_);
    float inv = rsqrtf(v + 1e-5f);
    for (int j = 0; j < T_; ++j) {
      float o = (row[j] - m) * inv * g1[j] + be1[j];
      row[j] = o;
      yh[tid * T_ + j] = (_Float16)o;
    }
  }
  __syncthreads();

  // ---- FF1: relu(y @ Wff1^T + bff1) -> ffh (64x1024) ----
  for (int t = w; t < 256; t += 8) {
    int ti = t >> 6, tj = t & 63;
    v8f c = {};
#pragma unroll
    for (int kk = 0; kk < 8; ++kk) {
      v16h a  = load_A16x32(yh + ti * 16 * T_ + kk * 32, T_, lane);
      v16h bm = load_B_NK(Wf1 + (size_t)(tj * 16) * T_ + kk * 32, T_, lane);
      c = wmma_f16(a, bm, c);
    }
    int n = lane & 15, m0 = (lane >> 4) << 3;
    int col = tj * 16 + n;
    float bias = bf1[col];
#pragma unroll
    for (int r = 0; r < 8; ++r)
      ffh[(size_t)(ti * 16 + m0 + r) * DFF_ + col] = (_Float16)fmaxf(c[r] + bias, 0.f);
  }
  __syncthreads();

  // ---- FF2 + residual: y32 += ffh @ Wff2^T + bff2 ----
  for (int t = w; t < 64; t += 8) {
    int ti = t >> 4, tj = t & 15;
    v8f c = {};
#pragma unroll 8
    for (int kk = 0; kk < 32; ++kk) {
      v16h a  = load_A16x32(ffh + (size_t)(ti * 16) * DFF_ + kk * 32, DFF_, lane);
      v16h bm = load_B_NK(Wf2 + (size_t)(tj * 16) * DFF_ + kk * 32, DFF_, lane);
      c = wmma_f16(a, bm, c);
    }
    int n = lane & 15, m0 = (lane >> 4) << 3;
    int col = tj * 16 + n;
    float bias = bf2[col];
#pragma unroll
    for (int r = 0; r < 8; ++r)
      y32[(ti * 16 + m0 + r) * T_ + col] += c[r] + bias;
  }
  __syncthreads();

  // ---- LayerNorm 2 ----
  if (tid < 64) {
    float* row = y32 + tid * T_;
    float m = 0.f;
    for (int j = 0; j < T_; ++j) m += row[j];
    m *= (1.f / T_);
    float v = 0.f;
    for (int j = 0; j < T_; ++j) { float d = row[j] - m; v += d * d; }
    v *= (1.f / T_);
    float inv = rsqrtf(v + 1e-5f);
    for (int j = 0; j < T_; ++j)
      row[j] = (row[j] - m) * inv * g2[j] + be2[j];
  }
  __syncthreads();

  // ---- writeback ----
  for (int i = tid; i < L_ * T_; i += 256) yg[i] = y32[i];
}

// ------------------------- value head --------------------------------------
// grid = B, block = 256
__global__ void head_kernel(const float* __restrict__ y,
                            const float* __restrict__ W1, const float* __restrict__ b1,
                            const float* __restrict__ W2, const float* __restrict__ b2,
                            float* __restrict__ out) {
  int b = blockIdx.x, t = threadIdx.x;
  __shared__ float pooled[T_];
  __shared__ float red[T_];
  float s = 0.f;
  for (int l = 0; l < L_; ++l) s += y[((size_t)b * L_ + l) * T_ + t];
  pooled[t] = s * (1.f / L_);
  __syncthreads();
  float acc = b1[t];
  for (int k = 0; k < T_; ++k) acc += pooled[k] * W1[(size_t)k * T_ + t];
  float hh = fmaxf(acc, 0.f);
  red[t] = hh * W2[t];
  __syncthreads();
  for (int s2 = 128; s2 > 0; s2 >>= 1) {
    if (t < s2) red[t] += red[t + s2];
    __syncthreads();
  }
  if (t == 0) out[b] = red[0] + b2[0];
}

// ------------------------- launch ------------------------------------------
extern "C" void kernel_launch(void* const* d_in, const int* in_sizes, int n_in,
                              void* d_out, int out_size, void* d_ws, size_t ws_size,
                              hipStream_t stream) {
  const float* link_states = (const float*)d_in[0];
  const float* link_W1 = (const float*)d_in[1];
  const float* link_b1 = (const float*)d_in[2];
  const float* link_W2 = (const float*)d_in[3];
  const float* link_b2 = (const float*)d_in[4];
  const float* Wqkv    = (const float*)d_in[5];
  const float* bqkv    = (const float*)d_in[6];
  const float* Wo      = (const float*)d_in[7];
  const float* bo      = (const float*)d_in[8];
  const float* ln1_g   = (const float*)d_in[9];
  const float* ln1_b   = (const float*)d_in[10];
  const float* Wff1    = (const float*)d_in[11];
  const float* bff1    = (const float*)d_in[12];
  const float* Wff2    = (const float*)d_in[13];
  const float* bff2    = (const float*)d_in[14];
  const float* ln2_g   = (const float*)d_in[15];
  const float* ln2_b   = (const float*)d_in[16];
  const float* head_W1 = (const float*)d_in[17];
  const float* head_b1 = (const float*)d_in[18];
  const float* head_W2 = (const float*)d_in[19];
  const float* head_b2 = (const float*)d_in[20];

  // workspace layout
  char* ws = (char*)d_ws;
  size_t off = 0;
  auto take = [&](size_t bytes) {
    size_t o = off;
    off += (bytes + 255) & ~(size_t)255;
    return o;
  };
  float*    xn    = (float*)(ws + take((size_t)B_ * L_ * DIN_ * 4));
  float*    y     = (float*)(ws + take((size_t)B_ * L_ * T_ * 4));
  _Float16* W2t   = (_Float16*)(ws + take((size_t)L_ * T_ * T_ * 2));
  _Float16* Wqkvh = (_Float16*)(ws + take((size_t)NL_ * 3 * T_ * T_ * 2));
  _Float16* Woh   = (_Float16*)(ws + take((size_t)NL_ * T_ * T_ * 2));
  _Float16* Wf1h  = (_Float16*)(ws + take((size_t)NL_ * DFF_ * T_ * 2));
  _Float16* Wf2h  = (_Float16*)(ws + take((size_t)NL_ * T_ * DFF_ * 2));
  (void)ws_size; (void)in_sizes; (void)n_in; (void)out_size;

  auto cvt = [&](const float* s, _Float16* d, size_t n) {
    cvt_f16_kernel<<<(unsigned)((n + 255) / 256), 256, 0, stream>>>(s, d, (int)n);
  };
  cvt_w2t_kernel<<<dim3(T_ * T_ / 256, L_), 256, 0, stream>>>(link_W2, W2t);
  cvt(Wqkv, Wqkvh, (size_t)NL_ * 3 * T_ * T_);
  cvt(Wo,   Woh,   (size_t)NL_ * T_ * T_);
  cvt(Wff1, Wf1h,  (size_t)NL_ * DFF_ * T_);
  cvt(Wff2, Wf2h,  (size_t)NL_ * T_ * DFF_);

  norm_kernel<<<B_, L_, 0, stream>>>(link_states, xn);

  link_kernel<<<dim3(B_ / 64, L_), 256, 0, stream>>>(xn, link_W1, link_b1, W2t,
                                                     link_b2, y);

  for (int l = 0; l < NL_; ++l) {
    layer_kernel<<<B_, 256, LAYER_SMEM, stream>>>(
        y,
        Wqkvh + (size_t)l * 3 * T_ * T_, bqkv + (size_t)l * 3 * T_,
        Woh   + (size_t)l * T_ * T_,     bo   + (size_t)l * T_,
        ln1_g + (size_t)l * T_,          ln1_b + (size_t)l * T_,
        Wf1h  + (size_t)l * DFF_ * T_,   bff1 + (size_t)l * DFF_,
        Wf2h  + (size_t)l * T_ * DFF_,   bff2 + (size_t)l * T_,
        ln2_g + (size_t)l * T_,          ln2_b + (size_t)l * T_);
  }

  head_kernel<<<B_, 256, 0, stream>>>(y, head_W1, head_b1, head_W2, head_b2,
                                      (float*)d_out);
}
